// GRU_28260884808404
// MI455X (gfx1250) — matrix operand order repaired and checked by reading, not compiled, not run on hardware
//
#include <hip/hip_runtime.h>

typedef unsigned short u16;
typedef unsigned int   u32;
typedef __attribute__((ext_vector_type(16))) __bf16 v16bf;
typedef __attribute__((ext_vector_type(8)))  float  v8f;
typedef __attribute__((ext_vector_type(4)))  u32    v4u;

#define SM_STRIDE 1032   // 1024 + 8 bf16 pad: kills LDS bank conflicts on b128 reads

// ---------- helpers ----------

static __device__ __forceinline__ u16 f2bf(float f) {
  u32 u = __builtin_bit_cast(u32, f);
  u = (u + 0x7FFFu + ((u >> 16) & 1u)) >> 16;   // round-to-nearest-even
  return (u16)u;
}

static __device__ __forceinline__ v8f wmma_bf16(v16bf a, v16bf b, v8f c) {
  // D(16x16 f32) = A(16x32 bf16) * B(32x16 bf16) + C
  return __builtin_amdgcn_wmma_f32_16x16x32_bf16(false, a, false, b, (short)0, c,
                                                 false, false);
}

union FragU { v16bf bf; v4u u[2]; };

// A fragment (16x32 bf16, M x K) from LDS, row-major, stride SM_STRIDE.
// lane<16: row=lane,   K chunks at k0+0..7  and k0+16..23
// lane>=16: row=lane-16, chunks at k0+8..15 and k0+24..31
static __device__ __forceinline__ v16bf loadA_lds(const u16* sm, int lane, int k0) {
  int row = lane & 15;
  int s   = (lane >> 4) << 3;            // 0 or 8
  const u16* p = sm + row * SM_STRIDE + k0 + s;
  FragU f;
  f.u[0] = *(const v4u*)(p);
  f.u[1] = *(const v4u*)(p + 16);
  return f.bf;
}

// B fragment (32x16 bf16, K x N) from global, weights pre-transposed to [N][K]:
// lanes 0-15 hold K=k0+0..7 and k0+16..23 of column N=lane, lanes 16-31 hold K=k0+8..15 and k0+24..31.
// -> two 16-byte chunks per lane.
static __device__ __forceinline__ v16bf loadB_g(const u16* wt, int n0, int lane, int k0) {
  int col = lane & 15;
  int s   = (lane >> 4) << 3;            // 0 or 8
  const u16* p = wt + (size_t)(n0 + col) * 1024 + k0 + s;
  FragU f;
  f.u[0] = *(const v4u*)(p);
  f.u[1] = *(const v4u*)(p + 16);
  return f.bf;
}

// Cooperative stage of a 16x1024 bf16 tile into LDS (256 threads).
static __device__ __forceinline__ void stage_bf16(const u16* src, u16* sm) {
  int t = threadIdx.x;
#pragma unroll
  for (int i = 0; i < 8; ++i) {
    int idx = t + i * 256;
    int row = idx >> 7;
    int col = (idx & 127) << 3;
    *(v4u*)(sm + row * SM_STRIDE + col) = *(const v4u*)(src + (size_t)row * 1024 + col);
  }
}

// Same, converting fp32 -> bf16 on the fly.
static __device__ __forceinline__ void stage_f32_to_bf16(const float* src, u16* sm) {
  int t = threadIdx.x;
#pragma unroll
  for (int i = 0; i < 8; ++i) {
    int idx = t + i * 256;
    int row = idx >> 7;
    int col = (idx & 127) << 3;
    const float* p = src + (size_t)row * 1024 + col;
    float4 f0 = *(const float4*)(p);
    float4 f1 = *(const float4*)(p + 4);
    v4u v;
    v.x = (u32)f2bf(f0.x) | ((u32)f2bf(f0.y) << 16);
    v.y = (u32)f2bf(f0.z) | ((u32)f2bf(f0.w) << 16);
    v.z = (u32)f2bf(f1.x) | ((u32)f2bf(f1.y) << 16);
    v.w = (u32)f2bf(f1.z) | ((u32)f2bf(f1.w) << 16);
    *(v4u*)(sm + row * SM_STRIDE + col) = v;
  }
}

// Device-wide sense barrier: arrive counter + generation.
// Relaxed spin (no per-poll cache invalidate) + one full fence after release.
static __device__ __forceinline__ void grid_sync(u32* bar, unsigned nblk) {
  __threadfence();           // flush this thread's global writes device-wide
  __syncthreads();           // whole block done + fenced before thread 0 arrives
  if (threadIdx.x == 0) {
    u32* cnt = bar + 0;
    u32* gen = bar + 1;
    u32 g = __hip_atomic_load(gen, __ATOMIC_ACQUIRE, __HIP_MEMORY_SCOPE_AGENT);
    u32 old = __hip_atomic_fetch_add(cnt, 1u, __ATOMIC_ACQ_REL, __HIP_MEMORY_SCOPE_AGENT);
    if (old == nblk - 1u) {
      __hip_atomic_store(cnt, 0u, __ATOMIC_RELAXED, __HIP_MEMORY_SCOPE_AGENT);
      __hip_atomic_store(gen, g + 1u, __ATOMIC_RELEASE, __HIP_MEMORY_SCOPE_AGENT);
    } else {
      while (__hip_atomic_load(gen, __ATOMIC_RELAXED, __HIP_MEMORY_SCOPE_AGENT) == g)
        __builtin_amdgcn_s_sleep(1);
      __threadfence();       // acquire: invalidate so we see other blocks' writes
    }
  }
  __syncthreads();
}

// ---------- kernels ----------

// Transpose+convert a 1024x1024 fp32 weight block (rows rowOff..rowOff+1023 of a
// [*,1024] matrix, stored [in][out]) into bf16 [N=out][K=in].
__global__ __launch_bounds__(256) void k_prep_w(const float* __restrict__ src,
                                                u16* __restrict__ dst, int rowOff) {
  int tid = blockIdx.x * 256 + threadIdx.x;   // 1M threads
  int n = tid >> 10;
  int k = tid & 1023;
  dst[(size_t)n * 1024 + k] = f2bf(src[(size_t)(rowOff + k) * 1024 + n]);
}

// Gx[m][0:3072] = X[m] @ [Wr_x|Wz_x|Wl_x] + [br|bz|bl],  m = b*512+t.
__global__ __launch_bounds__(256) void k_gemm_gx(
    const float* __restrict__ X,    // [32768][1024] fp32
    const u16*   __restrict__ WxT,  // [3072][1024] bf16 (pre-transposed)
    const float* __restrict__ br, const float* __restrict__ bz,
    const float* __restrict__ bl,
    float* __restrict__ Gx)         // [32768][3072] fp32
{
  __shared__ u16 smA[16 * SM_STRIDE];
  const int lane = threadIdx.x & 31;
  const int wave = threadIdx.x >> 5;
  const int m0 = blockIdx.y << 4;                         // 0..32752
  const int n0 = (blockIdx.x * 8 + wave) << 6;            // 0..3008, 16x64 per wave
  stage_f32_to_bf16(X + (size_t)m0 * 1024, smA);
  __syncthreads();
  v8f acc[4] = {};
  for (int k0 = 0; k0 < 1024; k0 += 32) {
    v16bf a = loadA_lds(smA, lane, k0);
#pragma unroll
    for (int j = 0; j < 4; ++j) {
      v16bf b = loadB_g(WxT, n0 + j * 16, lane, k0);
      acc[j] = wmma_bf16(a, b, acc[j]);
    }
  }
#pragma unroll
  for (int j = 0; j < 4; ++j) {
    int n = n0 + j * 16 + (lane & 15);
    int gate = n >> 10, nl = n & 1023;
    float bias = (gate == 0) ? br[nl] : (gate == 1) ? bz[nl] : bl[nl];
#pragma unroll
    for (int r = 0; r < 8; ++r) {
      int m = m0 + r + ((lane >> 4) << 3);
      Gx[(size_t)m * 3072 + n] = acc[j][r] + bias;
    }
  }
}

// Persistent recurrence: 64 blocks x 256 threads (512 waves), device barrier per phase.
// Each wave keeps TWO independent accumulators over split-K so back-to-back WMMAs
// are independent (hides XDL latency at low occupancy).
__global__ __launch_bounds__(256) void k_gru(
    const float* __restrict__ Gx,      // [32768][3072]
    const u16*   __restrict__ WhrzT,   // [2048][1024] bf16 ([Wr_h|Wz_h]^T)
    const u16*   __restrict__ WhlT,    // [1024][1024] bf16
    float* __restrict__ h_f32, float* __restrict__ z_f32,
    u16* __restrict__ h_bf16, u16* __restrict__ a_bf16,
    u16* __restrict__ Hist,            // [512][64][1024] bf16 (all h_t)
    u32* bar)
{
  __shared__ u16 smA[16 * SM_STRIDE];
  const int lane = threadIdx.x & 31;
  const int wave = threadIdx.x >> 5;
  const int blk  = blockIdx.x;                       // 0..63
  // h0 = 0
  for (int i = blk * 256 + threadIdx.x; i < 64 * 1024; i += 64 * 256) {
    h_f32[i] = 0.f; h_bf16[i] = 0;
  }
  grid_sync(bar, 64);

  const int mA = (blk >> 4) << 4;                    // phase A: 4 x 128 tiles
  const int nA = (((blk & 15) << 3) + wave) << 4;    // 0..2032
  const int mB = (blk >> 3) << 4;                    // phase B (blk<32): 4 x 64 tiles
  const int nB = (((blk & 7) << 3) + wave) << 4;     // 0..1008

  for (int t = 0; t < 512; ++t) {
    // ---- phase A: acc = h @ [Wr_h|Wz_h]; r -> a=h*r (bf16), z (fp32) ----
    stage_bf16(h_bf16 + (size_t)mA * 1024, smA);
    __syncthreads();
    v8f acc0 = {}, acc1 = {};
    for (int k0 = 0; k0 < 512; k0 += 32) {           // split-K: two indep chains
      v16bf a0 = loadA_lds(smA, lane, k0);
      v16bf b0 = loadB_g(WhrzT, nA, lane, k0);
      acc0 = wmma_bf16(a0, b0, acc0);
      v16bf a1 = loadA_lds(smA, lane, k0 + 512);
      v16bf b1 = loadB_g(WhrzT, nA, lane, k0 + 512);
      acc1 = wmma_bf16(a1, b1, acc1);
    }
    v8f acc = acc0 + acc1;
    const int n = nA + (lane & 15);
    if (nA < 1024) {
#pragma unroll
      for (int r = 0; r < 8; ++r) {
        int m = mA + r + ((lane >> 4) << 3);
        float g  = Gx[(size_t)(m * 512 + t) * 3072 + n];      // r-gate slot
        float rv = 1.f / (1.f + __expf(-(acc[r] + g)));
        a_bf16[m * 1024 + n] = f2bf(h_f32[m * 1024 + n] * rv);
      }
    } else {
#pragma unroll
      for (int r = 0; r < 8; ++r) {
        int m = mA + r + ((lane >> 4) << 3);
        float g = Gx[(size_t)(m * 512 + t) * 3072 + n];       // z-gate slot (n>=1024)
        z_f32[m * 1024 + (n - 1024)] = 1.f / (1.f + __expf(-(acc[r] + g)));
      }
    }
    grid_sync(bar, 64);

    // ---- phase B: hh = tanh((h*r) @ Wl_h + gx_l); h = (1-z)h + z*hh ----
    if (blk < 32) {
      stage_bf16(a_bf16 + (size_t)mB * 1024, smA);
      __syncthreads();
      v8f accB0 = {}, accB1 = {};
      for (int k0 = 0; k0 < 512; k0 += 32) {
        v16bf a0 = loadA_lds(smA, lane, k0);
        v16bf b0 = loadB_g(WhlT, nB, lane, k0);
        accB0 = wmma_bf16(a0, b0, accB0);
        v16bf a1 = loadA_lds(smA, lane, k0 + 512);
        v16bf b1 = loadB_g(WhlT, nB, lane, k0 + 512);
        accB1 = wmma_bf16(a1, b1, accB1);
      }
      v8f accB = accB0 + accB1;
      const int nb = nB + (lane & 15);
#pragma unroll
      for (int r = 0; r < 8; ++r) {
        int m = mB + r + ((lane >> 4) << 3);
        float g  = Gx[(size_t)(m * 512 + t) * 3072 + 2048 + nb];
        float v  = accB[r] + g;
        float e  = __expf(-2.f * v);
        float hh = (1.f - e) / (1.f + e);                     // tanh
        float z  = z_f32[m * 1024 + nb];
        float hn = (1.f - z) * h_f32[m * 1024 + nb] + z * hh;
        h_f32[m * 1024 + nb] = hn;
        u16 hb = f2bf(hn);
        h_bf16[m * 1024 + nb] = hb;
        Hist[((size_t)t * 64 + m) * 1024 + nb] = hb;
      }
    }
    grid_sync(bar, 64);
  }
}

// out[b][t][:] = relu(h_t[b] @ Wo + bo) as one big GEMM over Hist.
__global__ __launch_bounds__(256) void k_gemm_out(
    const u16* __restrict__ Hist,   // [32768][1024], row m = t*64+b
    const u16* __restrict__ WoT,    // [1024][1024] bf16
    const float* __restrict__ bo,
    float* __restrict__ out)        // [64][512][1024] fp32
{
  __shared__ u16 smA[16 * SM_STRIDE];
  const int lane = threadIdx.x & 31;
  const int wave = threadIdx.x >> 5;
  const int m0 = (blockIdx.x >> 1) << 4;
  const int n0 = (((blockIdx.x & 1) << 3) + wave) << 6;
  stage_bf16(Hist + (size_t)m0 * 1024, smA);
  __syncthreads();
  v8f acc[4] = {};
  for (int k0 = 0; k0 < 1024; k0 += 32) {
    v16bf a = loadA_lds(smA, lane, k0);
#pragma unroll
    for (int j = 0; j < 4; ++j) {
      v16bf b = loadB_g(WoT, n0 + j * 16, lane, k0);
      acc[j] = wmma_bf16(a, b, acc[j]);
    }
  }
#pragma unroll
  for (int j = 0; j < 4; ++j) {
    int n = n0 + j * 16 + (lane & 15);
    float bias = bo[n];
#pragma unroll
    for (int r = 0; r < 8; ++r) {
      int m = m0 + r + ((lane >> 4) << 3);
      int t = m >> 6, b = m & 63;
      out[(size_t)(b * 512 + t) * 1024 + n] = fmaxf(acc[j][r] + bias, 0.f);
    }
  }
}

// ---------- launch ----------

extern "C" void kernel_launch(void* const* d_in, const int* in_sizes, int n_in,
                              void* d_out, int out_size, void* d_ws, size_t ws_size,
                              hipStream_t stream) {
  (void)in_sizes; (void)n_in; (void)out_size; (void)ws_size;
  const float* X  = (const float*)d_in[0];
  const float* Wr = (const float*)d_in[1];
  const float* br = (const float*)d_in[2];
  const float* Wz = (const float*)d_in[3];
  const float* bz = (const float*)d_in[4];
  const float* Wl = (const float*)d_in[5];
  const float* bl = (const float*)d_in[6];
  const float* Wo = (const float*)d_in[7];
  const float* bo = (const float*)d_in[8];
  float* out = (float*)d_out;

  char* w = (char*)d_ws;
  u32*   bar    = (u32*)w;                                // 256 B barrier area
  float* h_f32  = (float*)(w + 256);                      // 64*1024 f32
  float* z_f32  = h_f32 + 64 * 1024;                      // 64*1024 f32
  u16*   h_bf16 = (u16*)(z_f32 + 64 * 1024);              // 64*1024 bf16
  u16*   a_bf16 = h_bf16 + 64 * 1024;                     // 64*1024 bf16
  u16*   WhrzT  = a_bf16 + 64 * 1024;                     // 2048*1024 bf16
  u16*   WhlT   = WhrzT + 2048 * 1024;                    // 1024*1024 bf16
  u16*   WoT    = WhlT + 1024 * 1024;                     // 1024*1024 bf16
  u16*   WxT    = WoT + 1024 * 1024;                      // 3072*1024 bf16
  u16*   Hist   = WxT + 3 * 1024 * 1024;                  // 512*64*1024 bf16
  float* Gx     = (float*)(Hist + (size_t)512 * 64 * 1024); // 32768*3072 f32

  hipMemsetAsync(bar, 0, 256, stream);

  dim3 blk(256);
  // bf16 transposed weight panels (h-parts, x-parts, Wo)
  k_prep_w<<<4096, blk, 0, stream>>>(Wr, WhrzT, 0);
  k_prep_w<<<4096, blk, 0, stream>>>(Wz, WhrzT + 1024 * 1024, 0);
  k_prep_w<<<4096, blk, 0, stream>>>(Wl, WhlT, 0);
  k_prep_w<<<4096, blk, 0, stream>>>(Wo, WoT, 0);
  k_prep_w<<<4096, blk, 0, stream>>>(Wr, WxT, 1024);
  k_prep_w<<<4096, blk, 0, stream>>>(Wz, WxT + 1024 * 1024, 1024);
  k_prep_w<<<4096, blk, 0, stream>>>(Wl, WxT + 2 * 1024 * 1024, 1024);
  // Parallel input projection for all timesteps (biases folded in)
  k_gemm_gx<<<dim3(6, 2048), blk, 0, stream>>>(X, WxT, br, bz, bl, Gx);
  // Sequential recurrence (persistent, device-wide barriers)
  k_gru<<<64, blk, 0, stream>>>(Gx, WhrzT, WhlT, h_f32, z_f32, h_bf16, a_bf16, Hist, bar);
  // Parallel output projection
  k_gemm_out<<<4096, blk, 0, stream>>>(Hist, WoT, bo, out);
}
